// TemporalGNN_6012954214766
// MI455X (gfx1250) — compile-verified, matrix operand order checked by reading, not run on hardware
//
#include <hip/hip_runtime.h>
#include <hip/hip_bf16.h>
#include <math.h>

typedef __bf16 bf16_t;
typedef bf16_t bf16x8  __attribute__((ext_vector_type(8)));
typedef bf16_t bf16x16 __attribute__((ext_vector_type(16)));
typedef float  f32x8   __attribute__((ext_vector_type(8)));

#define kN  50000
#define kE  800000
#define kFD 512
#define kHD 256
#define kHE 8
#define kC  32
#define kL  3
#define kG  64
#define kEPS 1e-5f

static __device__ __forceinline__ unsigned fenc(float f) {
  unsigned u = __float_as_uint(f);
  return (u & 0x80000000u) ? ~u : (u | 0x80000000u);
}
static __device__ __forceinline__ float fdec(unsigned u) {
  u = (u & 0x80000000u) ? (u & 0x7fffffffu) : ~u;
  return __uint_as_float(u);
}
static __device__ __forceinline__ float lrelu(float x) { return x > 0.f ? x : 0.2f * x; }

// ---------------- utility kernels ----------------

__global__ void k_fill_u32(unsigned* p, unsigned v, int n) {
  int t = blockIdx.x * blockDim.x + threadIdx.x;
  if (t < n) p[t] = v;
}

__global__ void k_cast_bf16(const float* __restrict__ a, bf16_t* __restrict__ b, int n) {
  int t = blockIdx.x * blockDim.x + threadIdx.x;
  if (t < n) b[t] = (bf16_t)a[t];
}

// W[K,N] fp32 row-major  ->  Wt[N,K] bf16 row-major (column-contiguous for WMMA B side)
__global__ void k_transpose_cast_bf16(const float* __restrict__ W, bf16_t* __restrict__ Wt,
                                      int K, int Nn) {
  int t = blockIdx.x * blockDim.x + threadIdx.x;
  if (t >= K * Nn) return;
  int k = t / Nn, n2 = t % Nn;
  Wt[(size_t)n2 * K + k] = (bf16_t)W[t];
}

// ---------------- WMMA GEMM: C[M,Nc] = A[M,K](bf16) * Bt[Nc,K]^T(bf16) + bias ----------------
// One wave per 16(M) x 64(N) strip: 4 f32 accumulators share one A fragment per K-step
// (4x A-reuse, 10 loads : 4 wmma in the inner loop). K-loop step 32, v_wmma_f32_16x16x32_bf16.
// Per-lane A: row m = tm*16+(lane&15); K chunks {k0..k0+7},{k0+16..k0+23}, k0 = kk + (lane>=16?8:0).
// Per-lane B: column n = nbase+j*16+(lane&15) from row-major Bt, same K pattern.
__global__ __launch_bounds__(256) void k_gemm_wmma(
    const bf16_t* __restrict__ A, const bf16_t* __restrict__ Bt,
    const float* __restrict__ bias, float* __restrict__ Cout,
    bf16_t* __restrict__ Cbf, int M, int K, int Nc, int relu) {
  const int wave = threadIdx.x >> 5;
  const int lane = threadIdx.x & 31;
  const int tilesN = Nc >> 6;             // 64-wide N strips
  const int tiles  = (M >> 4) * tilesN;
  const int t = blockIdx.x * 8 + wave;
  if (t >= tiles) return;                 // wave-uniform: EXEC stays all-ones for WMMA
  const int tm = t / tilesN, tn = t % tilesN;
  const int half = lane >> 4;
  const int l15  = lane & 15;
  const bf16_t* arow = A  + (size_t)(tm * 16 + l15) * K;
  const bf16_t* brow = Bt + (size_t)(tn * 64 + l15) * K;
  f32x8 acc[4];
#pragma unroll
  for (int j = 0; j < 4; j++) acc[j] = (f32x8){0.f, 0.f, 0.f, 0.f, 0.f, 0.f, 0.f, 0.f};
  for (int kk = 0; kk < K; kk += 32) {
    const int k0 = kk + half * 8;
    bf16x8 a_lo = *(const bf16x8*)(arow + k0);
    bf16x8 a_hi = *(const bf16x8*)(arow + k0 + 16);
    bf16x16 av;
#pragma unroll
    for (int i = 0; i < 8; i++) { av[i] = a_lo[i]; av[i + 8] = a_hi[i]; }
    if (kk + 32 < K)                       // keep-near prefetch of next K chunk
      __builtin_prefetch(arow + kk + 32, 0, 3);
#pragma unroll
    for (int j = 0; j < 4; j++) {
      const bf16_t* br = brow + (size_t)(j * 16) * K;
      bf16x8 b_lo = *(const bf16x8*)(br + k0);
      bf16x8 b_hi = *(const bf16x8*)(br + k0 + 16);
      bf16x16 bv;
#pragma unroll
      for (int i = 0; i < 8; i++) { bv[i] = b_lo[i]; bv[i + 8] = b_hi[i]; }
      acc[j] = __builtin_amdgcn_wmma_f32_16x16x32_bf16(
          false, av, false, bv, (short)0, acc[j], false, false);
    }
  }
  const int rbase = tm * 16 + half * 8;
#pragma unroll
  for (int j = 0; j < 4; j++) {
    const int n = tn * 64 + j * 16 + l15;
    const float bb = bias ? bias[n] : 0.f;
#pragma unroll
    for (int r = 0; r < 8; r++) {
      float v = acc[j][r] + bb;
      if (relu) v = fmaxf(v, 0.f);
      size_t idx = (size_t)(rbase + r) * Nc + n;
      Cout[idx] = v;
      if (Cbf) Cbf[idx] = (bf16_t)v;
    }
  }
}

// ---------------- attention coefficients: asn/adn [N,HE] ----------------

__global__ void k_attn(const float* __restrict__ xh, const float* __restrict__ a_src,
                       const float* __restrict__ a_dst, float* __restrict__ asn,
                       float* __restrict__ adn, int total) {
  int t = blockIdx.x * blockDim.x + threadIdx.x;
  if (t >= total) return;
  int h = t % kHE;
  const float* v = xh + (size_t)t * kC;   // t = n*HE+h, xh row = n*256 + h*32
  float s1 = 0.f, s2 = 0.f;
#pragma unroll
  for (int c = 0; c < kC; c++) {
    float x = v[c];
    s1 += x * a_src[h * kC + c];
    s2 += x * a_dst[h * kC + c];
  }
  asn[t] = s1;
  adn[t] = s2;
}

// ---------------- edge passes (E2 = E + N self-loops) ----------------

__global__ void k_edge_max(const int* __restrict__ ei, const float* __restrict__ asn,
                           const float* __restrict__ adn, unsigned* __restrict__ menc) {
  int t = blockIdx.x * blockDim.x + threadIdx.x;
  if (t >= (kE + kN) * kHE) return;
  int e = t / kHE, h = t % kHE;
  int s, d;
  if (e < kE) { s = ei[e]; d = ei[kE + e]; } else { s = d = e - kE; }
  float x = lrelu(asn[s * kHE + h] + adn[d * kHE + h]);
  atomicMax(&menc[d * kHE + h], fenc(x));
}

__global__ void k_edge_expsum(const int* __restrict__ ei, const float* __restrict__ asn,
                              const float* __restrict__ adn, const unsigned* __restrict__ menc,
                              float* __restrict__ den) {
  int t = blockIdx.x * blockDim.x + threadIdx.x;
  if (t >= (kE + kN) * kHE) return;
  int e = t / kHE, h = t % kHE;
  int s, d;
  if (e < kE) { s = ei[e]; d = ei[kE + e]; } else { s = d = e - kE; }
  int dh = d * kHE + h;
  float x = lrelu(asn[s * kHE + h] + adn[dh]);
  atomicAdd(&den[dh], expf(x - fdec(menc[dh])));
}

// one wave per (edge,head); lane = channel -> coalesced 128B gather + coalesced atomics
__global__ __launch_bounds__(256) void k_edge_msg(
    const int* __restrict__ ei, const float* __restrict__ xh,
    const float* __restrict__ asn, const float* __restrict__ adn,
    const unsigned* __restrict__ menc, const float* __restrict__ den,
    float* __restrict__ agg) {
  int wave = threadIdx.x >> 5, lane = threadIdx.x & 31;
  long long id = (long long)blockIdx.x * 8 + wave;
  if (id >= (long long)(kE + kN) * kHE) return;
  int e = (int)(id / kHE), h = (int)(id % kHE);
  int s, d;
  if (e < kE) { s = ei[e]; d = ei[kE + e]; } else { s = d = e - kE; }
  int dh = d * kHE + h;
  float x = lrelu(asn[s * kHE + h] + adn[dh]);
  float alpha = expf(x - fdec(menc[dh])) / (den[dh] + 1e-16f);
  float val = xh[(size_t)(s * kHE + h) * kC + lane] * alpha;
  atomicAdd(&agg[(size_t)dh * kC + lane], val);
}

// ---------------- residual + LayerNorm + ReLU (one 256-thread block per node) ----------------

__global__ __launch_bounds__(256) void k_ln(
    const float* __restrict__ agg, const float* __restrict__ hres,
    const float* __restrict__ gb, const float* __restrict__ g,
    const float* __restrict__ b, float* __restrict__ h, bf16_t* __restrict__ hbf) {
  int n = blockIdx.x, i = threadIdx.x;
  float v = agg[(size_t)n * kHD + i] + gb[i] + hres[(size_t)n * kHD + i];
  float s = v, s2 = v * v;
#pragma unroll
  for (int o = 16; o > 0; o >>= 1) { s += __shfl_down(s, o); s2 += __shfl_down(s2, o); }
  __shared__ float ss[8], ssq[8];
  int w = i >> 5, l = i & 31;
  if (l == 0) { ss[w] = s; ssq[w] = s2; }
  __syncthreads();
  if (i == 0) {
    float t = 0.f, t2 = 0.f;
    for (int j = 0; j < 8; j++) { t += ss[j]; t2 += ssq[j]; }
    float mu = t / kHD;
    ss[0] = mu;
    ssq[0] = t2 / kHD - mu * mu;
  }
  __syncthreads();
  float y = (v - ss[0]) * rsqrtf(ssq[0] + kEPS) * g[i] + b[i];
  y = fmaxf(y, 0.f);
  h[(size_t)n * kHD + i] = y;
  hbf[(size_t)n * kHD + i] = (bf16_t)y;
}

// ---------------- pooling ----------------

__global__ void k_pool_acc(const float* __restrict__ h, const int* __restrict__ batch,
                           float* __restrict__ sum, unsigned* __restrict__ mx,
                           float* __restrict__ cnt) {
  long long t = (long long)blockIdx.x * blockDim.x + threadIdx.x;
  if (t >= (long long)kN * kHD) return;
  int n = (int)(t / kHD), i = (int)(t % kHD);
  int g = batch[n];
  float v = h[t];
  atomicAdd(&sum[g * kHD + i], v);
  atomicMax(&mx[g * kHD + i], fenc(v));
  if (i == 0) atomicAdd(&cnt[g], 1.f);
}

__global__ void k_pool_fin(const float* __restrict__ sum, const unsigned* __restrict__ mx,
                           const float* __restrict__ cnt, bf16_t* __restrict__ hcat) {
  int t = blockIdx.x * blockDim.x + threadIdx.x;
  if (t >= kG * kHD) return;
  int g = t / kHD, i = t % kHD;
  float c = fmaxf(cnt[g], 1.f);
  hcat[(size_t)g * 2 * kHD + i] = (bf16_t)(sum[t] / c);
  hcat[(size_t)g * 2 * kHD + kHD + i] = (bf16_t)fdec(mx[t]);
}

// ---------------- host launcher ----------------

extern "C" void kernel_launch(void* const* d_in, const int* in_sizes, int n_in,
                              void* d_out, int out_size, void* d_ws, size_t ws_size,
                              hipStream_t stream) {
  (void)in_sizes; (void)n_in; (void)out_size; (void)ws_size;
  const float* x       = (const float*)d_in[0];
  const int*   ei      = (const int*)d_in[1];
  const int*   batch   = (const int*)d_in[2];
  const float* in_w    = (const float*)d_in[3];
  const float* in_b    = (const float*)d_in[4];
  const float* gat_w   = (const float*)d_in[5];
  const float* att_src = (const float*)d_in[6];
  const float* att_dst = (const float*)d_in[7];
  const float* gat_b   = (const float*)d_in[8];
  const float* res_w   = (const float*)d_in[9];
  const float* res_b   = (const float*)d_in[10];
  const float* ln_g    = (const float*)d_in[11];
  const float* ln_b    = (const float*)d_in[12];
  const float* pool_w  = (const float*)d_in[13];
  const float* pool_b  = (const float*)d_in[14];
  float* out = (float*)d_out;

  char* ws = (char*)d_ws;
  size_t off = 0;
  auto alloc = [&](size_t bytes) -> char* {
    char* p = ws + off;
    off += (bytes + 255) & ~(size_t)255;
    return p;
  };
  float*    h      = (float*)   alloc((size_t)kN * kHD * 4);
  bf16_t*   hbf    = (bf16_t*)  alloc((size_t)kN * kHD * 2);
  bf16_t*   xbf    = (bf16_t*)  alloc((size_t)kN * kFD * 2);
  float*    xh     = (float*)   alloc((size_t)kN * kHD * 4);
  float*    hres   = (float*)   alloc((size_t)kN * kHD * 4);
  float*    agg    = (float*)   alloc((size_t)kN * kHD * 4);
  float*    asn    = (float*)   alloc((size_t)kN * kHE * 4);
  float*    adn    = (float*)   alloc((size_t)kN * kHE * 4);
  unsigned* menc   = (unsigned*)alloc((size_t)kN * kHE * 4);
  float*    den    = (float*)   alloc((size_t)kN * kHE * 4);
  bf16_t*   inWt   = (bf16_t*)  alloc((size_t)kHD * kFD * 2);
  bf16_t*   gatWt  = (bf16_t*)  alloc((size_t)kL * kHD * kHD * 2);
  bf16_t*   resWt  = (bf16_t*)  alloc((size_t)kL * kHD * kHD * 2);
  bf16_t*   poolWt = (bf16_t*)  alloc((size_t)kHD * 2 * kHD * 2);
  float*    psum   = (float*)   alloc((size_t)kG * kHD * 4);
  unsigned* pmax   = (unsigned*)alloc((size_t)kG * kHD * 4);
  float*    pcnt   = (float*)   alloc((size_t)kG * 4);
  bf16_t*   hcat   = (bf16_t*)  alloc((size_t)kG * 2 * kHD * 2);

  auto nb = [](long long n, int b) { return (unsigned)((n + b - 1) / b); };

  // casts / weight pre-transposes (bf16, column-contiguous B)
  k_cast_bf16<<<nb((long long)kN * kFD, 256), 256, 0, stream>>>(x, xbf, kN * kFD);
  k_transpose_cast_bf16<<<nb(kFD * kHD, 256), 256, 0, stream>>>(in_w, inWt, kFD, kHD);
  for (int l = 0; l < kL; l++) {
    k_transpose_cast_bf16<<<nb(kHD * kHD, 256), 256, 0, stream>>>(
        gat_w + (size_t)l * kHD * kHD, gatWt + (size_t)l * kHD * kHD, kHD, kHD);
    k_transpose_cast_bf16<<<nb(kHD * kHD, 256), 256, 0, stream>>>(
        res_w + (size_t)l * kHD * kHD, resWt + (size_t)l * kHD * kHD, kHD, kHD);
  }
  k_transpose_cast_bf16<<<nb(2 * kHD * kHD, 256), 256, 0, stream>>>(pool_w, poolWt, 2 * kHD, kHD);

  // input projection: h = x @ in_w + in_b (WMMA bf16, fp32 accum)
  {
    int tiles = (kN / 16) * (kHD / 64);
    k_gemm_wmma<<<nb(tiles, 8), 256, 0, stream>>>(xbf, inWt, in_b, h, hbf, kN, kFD, kHD, 0);
  }

  const long long EH = (long long)(kE + kN) * kHE;
  for (int l = 0; l < kL; l++) {
    int tiles = (kN / 16) * (kHD / 64);
    k_gemm_wmma<<<nb(tiles, 8), 256, 0, stream>>>(
        hbf, resWt + (size_t)l * kHD * kHD, res_b + l * kHD, hres, nullptr, kN, kHD, kHD, 0);
    k_gemm_wmma<<<nb(tiles, 8), 256, 0, stream>>>(
        hbf, gatWt + (size_t)l * kHD * kHD, nullptr, xh, nullptr, kN, kHD, kHD, 0);
    k_attn<<<nb((long long)kN * kHE, 256), 256, 0, stream>>>(
        xh, att_src + l * kHE * kC, att_dst + l * kHE * kC, asn, adn, kN * kHE);
    k_fill_u32<<<nb(kN * kHE, 256), 256, 0, stream>>>(menc, 0u, kN * kHE);
    k_fill_u32<<<nb(kN * kHE, 256), 256, 0, stream>>>((unsigned*)den, 0u, kN * kHE);
    k_fill_u32<<<nb((long long)kN * kHD, 256), 256, 0, stream>>>((unsigned*)agg, 0u, kN * kHD);
    k_edge_max<<<nb(EH, 256), 256, 0, stream>>>(ei, asn, adn, menc);
    k_edge_expsum<<<nb(EH, 256), 256, 0, stream>>>(ei, asn, adn, menc, den);
    k_edge_msg<<<nb(EH, 8), 256, 0, stream>>>(ei, xh, asn, adn, menc, den, agg);
    k_ln<<<kN, kHD, 0, stream>>>(agg, hres, gat_b + l * kHD, ln_g + l * kHD, ln_b + l * kHD, h, hbf);
  }

  // pooling (mean + max) -> hcat[G, 2*HID] bf16 -> final WMMA GEMM with fused ReLU
  k_fill_u32<<<nb(kG * kHD, 256), 256, 0, stream>>>((unsigned*)psum, 0u, kG * kHD);
  k_fill_u32<<<nb(kG * kHD, 256), 256, 0, stream>>>(pmax, 0x00800000u /* fenc(-FLT_MAX) */, kG * kHD);
  k_fill_u32<<<nb(kG, 256), 256, 0, stream>>>((unsigned*)pcnt, 0u, kG);
  k_pool_acc<<<nb((long long)kN * kHD, 256), 256, 0, stream>>>(h, batch, psum, pmax, pcnt);
  k_pool_fin<<<nb(kG * kHD, 256), 256, 0, stream>>>(psum, pmax, pcnt, hcat);
  {
    int tiles = (kG / 16) * (kHD / 64);
    k_gemm_wmma<<<nb(tiles, 8), 256, 0, stream>>>(hcat, poolWt, pool_b, out, nullptr, kG, 2 * kHD, kHD, 1);
  }
}
